// WorldModel_9844065042634
// MI455X (gfx1250) — compile-verified, matrix operand order checked by reading, not run on hardware
//
#include <hip/hip_runtime.h>
#include <hip/hip_bf16.h>

// ---------------------------------------------------------------------------
// WorldModel forward for MI455X (gfx1250): bf16 WMMA GEMMs, f32 accumulate,
// async global->LDS DMA for activation tiles, transposed bf16 weight tiles.
// ---------------------------------------------------------------------------

typedef __attribute__((ext_vector_type(16))) __bf16 v16bf;
typedef __attribute__((ext_vector_type(8)))  __bf16 v8bf;
typedef __attribute__((ext_vector_type(8)))  float  v8f;

static constexpr int B_    = 4;
static constexpr int S_    = 1020;
static constexpr int D_    = 1024;
static constexpr int H_    = 16;
static constexpr int L_    = 4;
static constexpr int TPB_  = 17;
static constexpr int DH_   = 64;
static constexpr int BS_   = B_ * S_;     // 4080
static constexpr int D4_   = 4 * D_;      // 4096
static constexpr int VOBS_ = 4096;
static constexpr int VACT_ = 16;
static constexpr int NOBS_ = 960;         // obs-head rows per batch
static constexpr int NACT_ = 60;          // act positions per batch

__device__ inline v8f wmma_bf16(v16bf a, v16bf b, v8f c) {
  // v_wmma_f32_16x16x32_bf16: D = A(16x32) * B(32x16) + C(16x16 f32)
  return __builtin_amdgcn_wmma_f32_16x16x32_bf16(
      /*neg_a=*/false, a, /*neg_b=*/false, b,
      /*c_mod=*/(short)0, c, /*reuse_a=*/false, /*reuse_b=*/false);
}

// ---------------------------------------------------------------------------
// Embedding: x[b,s,:] = (obs? emb_obs[tok] : emb_act[tok]) + pos_emb[s]
// ---------------------------------------------------------------------------
__global__ __launch_bounds__(256)
void wm_embed(const int* __restrict__ tokens, const float* __restrict__ pos_emb,
              const float* __restrict__ emb_obs, const float* __restrict__ emb_act,
              float* __restrict__ x) {
  const int bs = blockIdx.x;                // 0..B*S-1
  const int sp = bs % S_;
  const bool is_obs = (sp % TPB_) < (TPB_ - 1);
  int tok = tokens[bs];
  const float* e;
  if (is_obs) {
    int t = tok < (VOBS_ - 1) ? tok : (VOBS_ - 1);
    e = emb_obs + (size_t)t * D_;
  } else {
    int t = tok < (VACT_ - 1) ? tok : (VACT_ - 1);
    e = emb_act + (size_t)t * D_;
  }
  float* xr = x + (size_t)bs * D_;
  const float* pr = pos_emb + (size_t)sp * D_;
  for (int c = threadIdx.x; c < D_; c += 256) xr[c] = e[c] + pr[c];
}

// ---------------------------------------------------------------------------
// LayerNorm over D=1024, one block per row.
// ---------------------------------------------------------------------------
__global__ __launch_bounds__(256)
void wm_ln(const float* __restrict__ x, const float* __restrict__ g,
           const float* __restrict__ b, float* __restrict__ out) {
  __shared__ float red[256];
  const int row = blockIdx.x;
  const int tid = threadIdx.x;
  const float* xr = x + (size_t)row * D_;
  float s = 0.f;
  for (int c = tid; c < D_; c += 256) s += xr[c];
  red[tid] = s; __syncthreads();
  for (int off = 128; off > 0; off >>= 1) {
    if (tid < off) red[tid] += red[tid + off];
    __syncthreads();
  }
  const float mean = red[0] * (1.0f / D_);
  __syncthreads();
  float v = 0.f;
  for (int c = tid; c < D_; c += 256) { float d = xr[c] - mean; v += d * d; }
  red[tid] = v; __syncthreads();
  for (int off = 128; off > 0; off >>= 1) {
    if (tid < off) red[tid] += red[tid + off];
    __syncthreads();
  }
  const float rstd = rsqrtf(red[0] * (1.0f / D_) + 1.0e-3f);
  float* orow = out + (size_t)row * D_;
  for (int c = tid; c < D_; c += 256)
    orow[c] = (xr[c] - mean) * rstd * g[c] + b[c];
}

// ---------------------------------------------------------------------------
// Generic WMMA GEMM: C[M,N] = act(A[M,K] * W[K,N] + bias) (+ residual R)
// Block tile 64x128, 8 waves (4 M-tiles x 2 N-slabs).
//  - A tile: f32, streamed global->LDS with GLOBAL_LOAD_ASYNC_TO_LDS_B128
//    (ASYNCcnt), converted to bf16 at fragment build.
//  - W tile: staged transposed [n][k] as bf16 so fragment reads are b128.
// ACT: 0=none, 1=gelu(exact), 2=relu
// ---------------------------------------------------------------------------
template <int ACT>
__global__ __launch_bounds__(256)
void wm_gemm(const float* __restrict__ A, const float* __restrict__ W,
             const float* __restrict__ bias, const float* __restrict__ R,
             float* __restrict__ C, int M, int N, int K) {
  __shared__ __attribute__((aligned(32))) float  Af[64][32];   // 8KB, async DMA
  __shared__ __attribute__((aligned(16))) __bf16 Bt[128][32];  // 8KB, transposed

  const int tid  = threadIdx.x;
  const int lane = tid & 31;
  const int w    = tid >> 5;
  const int wm   = w >> 1;          // 0..3
  const int wn   = w & 1;           // 0..1
  const int row0 = blockIdx.y * 64;
  const int col0 = blockIdx.x * 128;
  const int m16  = lane & 15;
  const int kh   = (lane >> 4) * 8;

  // Per-thread A-tile DMA slots: 512 x b128 chunks (4 floats), 2 per thread.
  const int ar0 = tid >> 3;               // 0..31
  const int ac0 = (tid & 7) * 4;
  const int ar1 = ar0 + 32;               // 32..63
  int gr0 = row0 + ar0; if (gr0 > M - 1) gr0 = M - 1;   // clamp; zeroed at use
  int gr1 = row0 + ar1; if (gr1 > M - 1) gr1 = M - 1;
  const unsigned lds0 = (unsigned)(unsigned long long)&Af[ar0][ac0];
  const unsigned lds1 = (unsigned)(unsigned long long)&Af[ar1][ac0];

  // W staging slot
  const int br = tid >> 3;                // k row 0..31
  const int bc = (tid & 7) * 16;          // n col 0,16,..,112

  v8f acc[4] = {};

  for (int k0 = 0; k0 < K; k0 += 32) {
    // Kick off async DMA of the A tile (f32) straight into LDS.
    {
      const float* g0 = A + (size_t)gr0 * K + (k0 + ac0);
      const float* g1 = A + (size_t)gr1 * K + (k0 + ac0);
      asm volatile("global_load_async_to_lds_b128 %0, %1, off"
                   :: "v"(lds0), "v"(g0) : "memory");
      asm volatile("global_load_async_to_lds_b128 %0, %1, off"
                   :: "v"(lds1), "v"(g1) : "memory");
    }
    // Prefetch next W chunk while we convert this one.
    if (k0 + 32 < K)
      __builtin_prefetch(W + (size_t)(k0 + 32 + br) * N + (col0 + bc), 0, 1);
    // Stage W (32x128) fp32 -> bf16, transposed to [n][k].
    {
      const float* src = W + (size_t)(k0 + br) * N + (col0 + bc);
      #pragma unroll
      for (int j = 0; j < 16; ++j) {
        const int gc = col0 + bc + j;
        Bt[bc + j][br] = (gc < N) ? (__bf16)src[j] : (__bf16)0.0f;
      }
    }
    // My DMAs done; barrier makes every wave's DMA + staging visible.
    asm volatile("s_wait_asynccnt 0" ::: "memory");
    __syncthreads();

    // A fragment: two aligned v8f reads from LDS, cvt to bf16.
    v16bf a;
    const int arow = wm * 16 + m16;
    if (row0 + arow < M) {
      const v8f a0 = *(const v8f*)&Af[arow][kh];
      const v8f a1 = *(const v8f*)&Af[arow][16 + kh];
      #pragma unroll
      for (int j = 0; j < 8; ++j) { a[j] = (__bf16)a0[j]; a[8 + j] = (__bf16)a1[j]; }
    } else {
      #pragma unroll
      for (int j = 0; j < 16; ++j) a[j] = (__bf16)0.0f;
    }
    // B fragments: contiguous b128 LDS reads from the transposed tile.
    #pragma unroll
    for (int t = 0; t < 4; ++t) {
      const int c = wn * 64 + t * 16 + m16;
      const v8bf lo = *(const v8bf*)&Bt[c][kh];
      const v8bf hi = *(const v8bf*)&Bt[c][16 + kh];
      const v16bf b = __builtin_shufflevector(lo, hi,
          0, 1, 2, 3, 4, 5, 6, 7, 8, 9, 10, 11, 12, 13, 14, 15);
      acc[t] = wmma_bf16(a, b, acc[t]);
    }
    __syncthreads();
  }

  #pragma unroll
  for (int t = 0; t < 4; ++t) {
    #pragma unroll
    for (int r = 0; r < 8; ++r) {
      const int gr = row0 + wm * 16 + ((lane >> 4) * 8) + r;
      const int gc = col0 + wn * 64 + t * 16 + m16;
      if (gr < M && gc < N) {
        float v = acc[t][r] + bias[gc];
        if (ACT == 1) v = 0.5f * v * (1.0f + erff(v * 0.70710678118654752f));
        else if (ACT == 2) v = fmaxf(v, 0.0f);
        if (R) v += R[(size_t)gr * N + gc];
        C[(size_t)gr * N + gc] = v;
      }
    }
  }
}

// ---------------------------------------------------------------------------
// Attention: one block per (b, h, 16-query tile). Full key row (pad 1024) in
// LDS, WMMA for Q*K^T and P*V, causal mask -1e9, softmax, 1/sum at store.
// ---------------------------------------------------------------------------
__global__ __launch_bounds__(256)
void wm_attn(const float* __restrict__ q, const float* __restrict__ k,
             const float* __restrict__ v, float* __restrict__ o) {
  __shared__ float sc[16][1024];    // scores -> probs; reused as partial-O
  __shared__ float rred[16][16];
  __shared__ float rmax[16];
  __shared__ float rsum[16];

  const int tid  = threadIdx.x;
  const int lane = tid & 31;
  const int w    = tid >> 5;        // 0..7
  const int qb   = blockIdx.x;      // 0..63
  const int hh   = blockIdx.y;      // 0..15
  const int bb   = blockIdx.z;      // 0..3
  const int q0   = qb * 16;
  const int ch   = hh * DH_;
  const int m16  = lane & 15;
  const int kh   = (lane >> 4) * 8;
  const size_t base = (size_t)bb * S_ * D_;

  // Q fragments (dh 0..31 and 32..63)
  v16bf aq0, aq1;
  {
    const int qr = q0 + m16;
    if (qr < S_) {
      const float* qp = q + base + (size_t)qr * D_ + ch;
      #pragma unroll
      for (int j = 0; j < 8; ++j) {
        aq0[j]     = (__bf16)qp[kh + j];
        aq0[8 + j] = (__bf16)qp[16 + kh + j];
        aq1[j]     = (__bf16)qp[32 + kh + j];
        aq1[8 + j] = (__bf16)qp[48 + kh + j];
      }
    } else {
      #pragma unroll
      for (int j = 0; j < 16; ++j) { aq0[j] = (__bf16)0.0f; aq1[j] = (__bf16)0.0f; }
    }
  }

  // Phase 1: scores = Q*K^T * 1/8 + mask
  for (int kt = w; kt < 64; kt += 8) {
    const int key = kt * 16 + m16;
    v16bf b0, b1;
    if (key < S_) {
      const float* kp = k + base + (size_t)key * D_ + ch;
      #pragma unroll
      for (int j = 0; j < 8; ++j) {
        b0[j]     = (__bf16)kp[kh + j];
        b0[8 + j] = (__bf16)kp[16 + kh + j];
        b1[j]     = (__bf16)kp[32 + kh + j];
        b1[8 + j] = (__bf16)kp[48 + kh + j];
      }
    } else {
      #pragma unroll
      for (int j = 0; j < 16; ++j) { b0[j] = (__bf16)0.0f; b1[j] = (__bf16)0.0f; }
    }
    v8f acc = {};
    acc = wmma_bf16(aq0, b0, acc);
    acc = wmma_bf16(aq1, b1, acc);
    #pragma unroll
    for (int r = 0; r < 8; ++r) {
      const int qm   = ((lane >> 4) * 8) + r;
      const int qpos = q0 + qm;
      const int kc   = kt * 16 + m16;
      float val = acc[r] * 0.125f;            // 1/sqrt(64)
      if (kc >= S_)       val = -3.0e38f;     // padding keys -> P=0
      else if (kc > qpos) val += -1.0e9f;     // causal mask (matches reference)
      if (qpos >= S_)     val = 0.0f;         // dead query rows: keep finite
      sc[qm][kt * 16 + m16] = val;
    }
  }
  __syncthreads();

  // Phase 2: row softmax (statistics only; normalize at the end)
  const int srow = tid >> 4;
  const int sl   = tid & 15;
  float mx = -3.0e38f;
  for (int c = sl; c < 1024; c += 16) mx = fmaxf(mx, sc[srow][c]);
  rred[srow][sl] = mx;
  __syncthreads();
  if (sl == 0) {
    float m2 = rred[srow][0];
    #pragma unroll
    for (int j = 1; j < 16; ++j) m2 = fmaxf(m2, rred[srow][j]);
    rmax[srow] = m2;
  }
  __syncthreads();
  const float rm = rmax[srow];
  float sum = 0.f;
  for (int c = sl; c < 1024; c += 16) {
    const float e = expf(sc[srow][c] - rm);
    sc[srow][c] = e;
    sum += e;
  }
  rred[srow][sl] = sum;
  __syncthreads();
  if (sl == 0) {
    float s2 = 0.f;
    #pragma unroll
    for (int j = 0; j < 16; ++j) s2 += rred[srow][j];
    rsum[srow] = s2;
  }
  __syncthreads();

  // Phase 3: O = P * V, contraction over 1024 padded keys, 4 chunks/wave
  v8f oacc[4] = {};
  for (int kc = w; kc < 32; kc += 8) {
    const int key0 = kc * 32;
    v16bf ap;
    #pragma unroll
    for (int j = 0; j < 8; ++j) {
      ap[j]     = (__bf16)sc[m16][key0 + kh + j];
      ap[8 + j] = (__bf16)sc[m16][key0 + 16 + kh + j];
    }
    #pragma unroll
    for (int t = 0; t < 4; ++t) {
      v16bf bv;
      const int n = t * 16 + m16;
      #pragma unroll
      for (int j = 0; j < 8; ++j) {
        const int k1 = key0 + kh + j;
        const int k2 = key0 + 16 + kh + j;
        bv[j]     = (k1 < S_) ? (__bf16)v[base + (size_t)k1 * D_ + ch + n] : (__bf16)0.0f;
        bv[8 + j] = (k2 < S_) ? (__bf16)v[base + (size_t)k2 * D_ + ch + n] : (__bf16)0.0f;
      }
      oacc[t] = wmma_bf16(ap, bv, oacc[t]);
    }
  }
  __syncthreads();  // all waves done reading sc

  // cross-wave reduce via LDS (reuse sc storage: 8*16*64 floats = 32KB)
  float* po = &sc[0][0];
  #pragma unroll
  for (int t = 0; t < 4; ++t) {
    #pragma unroll
    for (int r = 0; r < 8; ++r) {
      const int qm = ((lane >> 4) * 8) + r;
      const int n  = t * 16 + m16;
      po[(w * 16 + qm) * 64 + n] = oacc[t][r];
    }
  }
  __syncthreads();
  for (int idx = tid; idx < 16 * 64; idx += 256) {
    const int qm = idx >> 6;
    const int n  = idx & 63;
    const int qr = q0 + qm;
    if (qr < S_) {
      float s2 = 0.f;
      #pragma unroll
      for (int ww = 0; ww < 8; ++ww) s2 += po[(ww * 16 + qm) * 64 + n];
      o[base + (size_t)qr * D_ + ch + n] = s2 / rsum[qm];
    }
  }
}

// ---------------------------------------------------------------------------
// Head gathers: obs rows (p%17 != 15) and act rows (p%17 == 16)
// ---------------------------------------------------------------------------
__global__ __launch_bounds__(256)
void wm_gather_obs(const float* __restrict__ xf, float* __restrict__ xo) {
  const int r = blockIdx.x;               // 0..B*960-1
  const int b = r / NOBS_, j = r % NOBS_;
  const int jm = j & 15;
  const int p = 17 * (j >> 4) + (jm == 15 ? 16 : jm);
  const float* src = xf + ((size_t)b * S_ + p) * D_;
  float* dst = xo + (size_t)r * D_;
  for (int c = threadIdx.x; c < D_; c += 256) dst[c] = src[c];
}

__global__ __launch_bounds__(256)
void wm_gather_ends(const float* __restrict__ xf, float* __restrict__ xe) {
  const int r = blockIdx.x;               // 0..B*60-1
  const int b = r / NACT_, j = r % NACT_;
  const int p = 17 * j + 16;
  const float* src = xf + ((size_t)b * S_ + p) * D_;
  float* dst = xe + (size_t)r * D_;
  for (int c = threadIdx.x; c < D_; c += 256) dst[c] = src[c];
}

// Tiny ends head: [240,1024] x [1024,2] + bias
__global__ __launch_bounds__(256)
void wm_ends_head(const float* __restrict__ he, const float* __restrict__ w2,
                  const float* __restrict__ bias, float* __restrict__ out) {
  __shared__ float red[256][2];
  const int r = blockIdx.x;
  const int tid = threadIdx.x;
  const float* hr = he + (size_t)r * D_;
  float s0 = 0.f, s1 = 0.f;
  for (int c = tid; c < D_; c += 256) {
    const float hv = hr[c];
    s0 += hv * w2[c * 2 + 0];
    s1 += hv * w2[c * 2 + 1];
  }
  red[tid][0] = s0; red[tid][1] = s1; __syncthreads();
  for (int off = 128; off > 0; off >>= 1) {
    if (tid < off) { red[tid][0] += red[tid + off][0]; red[tid][1] += red[tid + off][1]; }
    __syncthreads();
  }
  if (tid < 2) out[(size_t)r * 2 + tid] = red[0][tid] + bias[tid];
}

// ---------------------------------------------------------------------------
// Orchestration
// ---------------------------------------------------------------------------
extern "C" void kernel_launch(void* const* d_in, const int* in_sizes, int n_in,
                              void* d_out, int out_size, void* d_ws, size_t ws_size,
                              hipStream_t stream) {
  const int*   tokens  = (const int*)  d_in[0];
  const float* pos_emb = (const float*)d_in[1];
  const float* emb_obs = (const float*)d_in[2];
  const float* emb_act = (const float*)d_in[3];
  const float* ln1_g   = (const float*)d_in[4];
  const float* ln1_b   = (const float*)d_in[5];
  const float* wq      = (const float*)d_in[6];
  const float* bq      = (const float*)d_in[7];
  const float* wk      = (const float*)d_in[8];
  const float* bk      = (const float*)d_in[9];
  const float* wv      = (const float*)d_in[10];
  const float* bv      = (const float*)d_in[11];
  const float* wo      = (const float*)d_in[12];
  const float* bo      = (const float*)d_in[13];
  const float* ln2_g   = (const float*)d_in[14];
  const float* ln2_b   = (const float*)d_in[15];
  const float* w1      = (const float*)d_in[16];
  const float* b1      = (const float*)d_in[17];
  const float* w2      = (const float*)d_in[18];
  const float* b2      = (const float*)d_in[19];
  const float* lnf_g   = (const float*)d_in[20];
  const float* lnf_b   = (const float*)d_in[21];
  const float* ho1     = (const float*)d_in[22];
  const float* bo1     = (const float*)d_in[23];
  const float* ho2     = (const float*)d_in[24];
  const float* bo2     = (const float*)d_in[25];
  const float* he1     = (const float*)d_in[26];
  const float* be1     = (const float*)d_in[27];
  const float* he2     = (const float*)d_in[28];
  const float* be2     = (const float*)d_in[29];
  float* out = (float*)d_out;

  float* ws = (float*)d_ws;
  size_t off = 0;
  auto alloc = [&](size_t n) { float* p = ws + off; off += n; return p; };
  float* x   = alloc((size_t)BS_ * D_);
  float* hb  = alloc((size_t)BS_ * D_);
  float* qb  = alloc((size_t)BS_ * D_);
  float* kb  = alloc((size_t)BS_ * D_);
  float* vb  = alloc((size_t)BS_ * D_);
  float* ob  = alloc((size_t)BS_ * D_);
  float* m1  = alloc((size_t)BS_ * D4_);
  float* xo  = alloc((size_t)B_ * NOBS_ * D_);
  float* hob = alloc((size_t)B_ * NOBS_ * D_);
  float* xe  = alloc((size_t)B_ * NACT_ * D_);
  float* heb = alloc((size_t)B_ * NACT_ * D_);
  (void)ws_size; (void)in_sizes; (void)n_in; (void)out_size;

  wm_embed<<<BS_, 256, 0, stream>>>(tokens, pos_emb, emb_obs, emb_act, x);

  const dim3 gD(D_ / 128, (BS_ + 63) / 64);     // N=1024 GEMMs over all tokens
  const dim3 gD4(D4_ / 128, (BS_ + 63) / 64);   // N=4096 GEMM

  for (int l = 0; l < L_; ++l) {
    const size_t dd = (size_t)l * D_ * D_;
    const size_t d1 = (size_t)l * D_;
    const size_t d4d = (size_t)l * D_ * D4_;
    const size_t d41 = (size_t)l * D4_;

    wm_ln<<<BS_, 256, 0, stream>>>(x, ln1_g + d1, ln1_b + d1, hb);
    wm_gemm<0><<<gD, 256, 0, stream>>>(hb, wq + dd, bq + d1, nullptr, qb, BS_, D_, D_);
    wm_gemm<0><<<gD, 256, 0, stream>>>(hb, wk + dd, bk + d1, nullptr, kb, BS_, D_, D_);
    wm_gemm<0><<<gD, 256, 0, stream>>>(hb, wv + dd, bv + d1, nullptr, vb, BS_, D_, D_);

    wm_attn<<<dim3(64, H_, B_), 256, 0, stream>>>(qb, kb, vb, ob);

    // x = x + o @ wo + bo
    wm_gemm<0><<<gD, 256, 0, stream>>>(ob, wo + dd, bo + d1, x, x, BS_, D_, D_);

    wm_ln<<<BS_, 256, 0, stream>>>(x, ln2_g + d1, ln2_b + d1, hb);
    wm_gemm<1><<<gD4, 256, 0, stream>>>(hb, w1 + d4d, b1 + d41, nullptr, m1, BS_, D4_, D_);
    // x = x + gelu_out @ w2 + b2
    wm_gemm<0><<<gD, 256, 0, stream>>>(m1, w2 + d4d, b2 + d1, x, x, BS_, D_, D4_);
  }

  wm_ln<<<BS_, 256, 0, stream>>>(x, lnf_g, lnf_b, hb);

  // Obs head: relu(xo @ ho1 + bo1) @ ho2 + bo2  -> out[0 : B*960*4096]
  const int MO = B_ * NOBS_;  // 3840
  wm_gather_obs<<<MO, 256, 0, stream>>>(hb, xo);
  wm_gemm<2><<<dim3(D_ / 128, MO / 64), 256, 0, stream>>>(xo, ho1, bo1, nullptr, hob, MO, D_, D_);
  wm_gemm<0><<<dim3(VOBS_ / 128, MO / 64), 256, 0, stream>>>(hob, ho2, bo2, nullptr, out, MO, VOBS_, D_);

  // Ends head: relu(xe @ he1 + be1) @ he2 + be2 -> out[B*960*4096 : +B*60*2]
  const int ME = B_ * NACT_;  // 240
  wm_gather_ends<<<ME, 256, 0, stream>>>(hb, xe);
  wm_gemm<2><<<dim3(D_ / 128, (ME + 63) / 64), 256, 0, stream>>>(xe, he1, be1, nullptr, heb, ME, D_, D_);
  wm_ends_head<<<ME, 256, 0, stream>>>(heb, he2, be2, out + (size_t)MO * VOBS_);
}
